// InnerSelfAttention_3135326126156
// MI455X (gfx1250) — matrix operand order repaired and checked by reading, not compiled, not run on hardware
//
#include <hip/hip_runtime.h>
#include <hip/hip_bf16.h>

// ---------------- problem constants ----------------
#define BATCH 4
#define NSEQ  1024
#define HID   1024
#define NH    16
#define DH    64
#define BH    (BATCH*NH)          // 64 (b,h) pairs

// ---------------- d_out layout (floats) ----------------
// tuple: (attention_values, k, q, context, scores) flattened in order
#define AV_OFF   0                 // 4*16*1024            = 65536
#define K_OFF    65536             // 4*16*1024*64         = 4194304
#define Q_OFF    4259840
#define C_OFF    8454144
#define S_OFF    12648448          // 4*16*1024*1024       = 67108864

// ---------------- workspace layout ----------------
// halves:
#define X16_OFF   0                 // 4096*1024      = 4194304 halves
#define W16_OFF   4194304           // 3*1024*1024    = 3145728 halves
#define Q16_OFF   7340032           // 4194304 halves
#define K16_OFF   11534336          // 4194304 halves
#define V16T_OFF  15728640          // 4194304 halves (per-head TRANSPOSED: [bh][d][n])
// bytes:
#define PADD_BYTE_OFF 39845888      // 4 ints
#define AVP_BYTE_OFF  39845952      // 4096*1024 floats partial column sums

typedef __attribute__((ext_vector_type(16))) _Float16 v16h;
typedef __attribute__((ext_vector_type(8)))  _Float16 v8h;
typedef __attribute__((ext_vector_type(8)))  float    v8f;

__device__ __forceinline__ v16h cat8(v8h lo, v8h hi) {
    v16h r;
#pragma unroll
    for (int i = 0; i < 8; ++i) { r[i] = lo[i]; r[i + 8] = hi[i]; }
    return r;
}

// ---------------------------------------------------------------------------
// Kernel 1: f32 -> f16 conversion of X and Wq/Wk/Wv (concatenated W16 buffer)
// total elems = 4194304 + 3145728 = 7340032 = 28672 blocks * 256
// ---------------------------------------------------------------------------
__global__ void __launch_bounds__(256) k_convert(
    const float* __restrict__ X, const float* __restrict__ Wq,
    const float* __restrict__ Wk, const float* __restrict__ Wv,
    _Float16* __restrict__ X16, _Float16* __restrict__ W16) {
    int i = blockIdx.x * 256 + threadIdx.x;
    if (i < 4194304) {
        X16[i] = (_Float16)X[i];
    } else {
        int w = i - 4194304;
        const float* src; int off;
        if (w < 1048576)      { src = Wq; off = w; }
        else if (w < 2097152) { src = Wk; off = w - 1048576; }
        else                  { src = Wv; off = w - 2097152; }
        W16[w] = (_Float16)src[off];
    }
}

// ---------------------------------------------------------------------------
// Kernel 2: padd[b] = index of first zero in padding_vector row, else N
// ---------------------------------------------------------------------------
__global__ void __launch_bounds__(256) k_padd(const int* __restrict__ pv,
                                              int* __restrict__ padd) {
    __shared__ int smin;
    int b = blockIdx.x;
    if (threadIdx.x == 0) smin = NSEQ;
    __syncthreads();
    int local = NSEQ;
    for (int m = threadIdx.x; m < NSEQ; m += 256)
        if (pv[b * NSEQ + m] == 0) local = min(local, m);
    atomicMin(&smin, local);
    __syncthreads();
    if (threadIdx.x == 0) padd[b] = smin;
}

// ---------------------------------------------------------------------------
// Kernel 3: QKV projection  C = X @ W^T + b.
// One 16x16 output tile per wave, K-loop over 1024 in steps of 32 (WMMA f16).
// grid = (2048, 3): blockIdx.y selects q/k/v. 8 waves per block.
// q,k: f32 (nontemporal, pure output) + f16 copy to ws (temporal, re-read).
// v: f16 transposed per head to ws.
// ---------------------------------------------------------------------------
__global__ void __launch_bounds__(256) k_qkv(
    const _Float16* __restrict__ X16, const _Float16* __restrict__ W16,
    const float* __restrict__ bq, const float* __restrict__ bk,
    const float* __restrict__ bv, float* __restrict__ dout,
    _Float16* __restrict__ q16, _Float16* __restrict__ k16,
    _Float16* __restrict__ v16t) {
    const int lane = threadIdx.x & 31;
    const int w    = threadIdx.x >> 5;
    const int half = lane >> 4;          // K-split selector
    const int rr   = lane & 15;          // row (A) / column (B/C) index
    const int mat  = blockIdx.y;         // 0=q 1=k 2=v
    const int tile = blockIdx.x * 8 + w; // 0..16383
    const int m0   = (tile >> 6) * 16;   // output row block (of 4096)
    const int o0   = (tile & 63) * 16;   // output col block (of 1024)

    const _Float16* Wm   = W16 + mat * 1048576;
    const _Float16* arow = X16 + (m0 + rr) * HID;   // A: row m0+rr for all lanes
    const _Float16* brow = Wm  + (o0 + rr) * HID;   // B: column o0+rr (W row-major)

    v8f acc = {};
#pragma unroll 4
    for (int k0 = 0; k0 < HID; k0 += 32) {
        v8h alo = *(const v8h*)(arow + k0 + 8 * half);        // K 0..7 / 8..15
        v8h ahi = *(const v8h*)(arow + k0 + 16 + 8 * half);   // K 16..23 / 24..31
        v16h a  = cat8(alo, ahi);
        v16h bm = cat8(*(const v8h*)(brow + k0 + 8 * half),   // K 0..7,16..23 / 8..15,24..31
                       *(const v8h*)(brow + k0 + 16 + 8 * half));
        acc = __builtin_amdgcn_wmma_f32_16x16x32_f16(false, a, false, bm,
                                                     (short)0, acc, false, false);
    }

    const int o  = o0 + rr;
    const float* bias = (mat == 0) ? bq : (mat == 1) ? bk : bv;
    const float bval  = bias[o];
    const int hh = o >> 6, dc = o & 63;

#pragma unroll
    for (int i = 0; i < 8; ++i) {
        int row = m0 + 8 * half + i;           // 0..4095 = (b,n)
        int bi  = row >> 10, n = row & 1023;
        int bh  = bi * NH + hh;
        float val = acc[i] + bval;
        int idx = (bh * NSEQ + n) * DH + dc;   // (B,h,N,d) flat
        if (mat == 0) {
            __builtin_nontemporal_store(val, dout + Q_OFF + idx);  // pure output
            q16[idx] = (_Float16)val;                              // re-read: RT
        } else if (mat == 1) {
            __builtin_nontemporal_store(val, dout + K_OFF + idx);
            k16[idx] = (_Float16)val;
        } else {
            v16t[(bh * DH + dc) * NSEQ + n] = (_Float16)val;
        }
    }
}

// ---------------------------------------------------------------------------
// Kernel 4: FUSED  scores (WMMA) -> masked softmax -> context (WMMA) + av.
// Block = 16 query rows of one (b,h) head; 4096 blocks, 8 waves.
//   Phase A: each wave computes 8 key-tiles (16x16) of the score strip with
//            the Q-strip A-fragments held in registers; raw scores stored
//            with regular (temporal) stores so the block's own 64 KB window
//            stays L2-resident for the immediate re-read.
//   Phase B: wave32 shuffle-reduction softmax (2 rows/wave), re-reading the
//            strip (L2 hit), masked per padd; probs -> LDS as f16.
//   Phase C: waves 0..3 WMMA probs @ V (ds_load_b128 A-frags, transposed V);
//            context via nontemporal stores. Deterministic av partials to ws.
// ---------------------------------------------------------------------------
__global__ void __launch_bounds__(256) k_sattn(
    const _Float16* __restrict__ q16, const _Float16* __restrict__ k16,
    const _Float16* __restrict__ v16t, const int* __restrict__ padd,
    float* __restrict__ dout, float* __restrict__ avp) {
    __shared__ _Float16 probs[16][NSEQ];     // 32 KB

    const int blk = blockIdx.x;              // 0..4095
    const int bh  = blk >> 6;
    const int n0  = (blk & 63) * 16;
    const int pd  = padd[bh >> 4];
    const int lane = threadIdx.x & 31;
    const int w    = threadIdx.x >> 5;
    const int half = lane >> 4;
    const int rr   = lane & 15;
    float* sbase = dout + S_OFF + (size_t)bh * NSEQ * NSEQ;

    // --- Phase A: scores strip = Q(16 x 64) K^T / 8, wave w owns 8 m-tiles ---
    {
        const _Float16* arow = q16 + (size_t)bh * NSEQ * DH + (n0 + rr) * DH;
        const v16h a0 = cat8(*(const v8h*)(arow + 8 * half),
                             *(const v8h*)(arow + 16 + 8 * half));
        const v16h a1 = cat8(*(const v8h*)(arow + 32 + 8 * half),
                             *(const v8h*)(arow + 48 + 8 * half));
        for (int mt = w * 8; mt < w * 8 + 8; ++mt) {
            const int m0 = mt * 16;
            const _Float16* brow = k16 + (size_t)bh * NSEQ * DH + (m0 + rr) * DH;
            v16h b0 = cat8(*(const v8h*)(brow + 8 * half),
                           *(const v8h*)(brow + 16 + 8 * half));
            v16h b1 = cat8(*(const v8h*)(brow + 32 + 8 * half),
                           *(const v8h*)(brow + 48 + 8 * half));
            v8f acc = {};
            acc = __builtin_amdgcn_wmma_f32_16x16x32_f16(false, a0, false, b0,
                                                         (short)0, acc, false, false);
            acc = __builtin_amdgcn_wmma_f32_16x16x32_f16(false, a1, false, b1,
                                                         (short)0, acc, false, false);
#pragma unroll
            for (int i = 0; i < 8; ++i)
                sbase[(size_t)(n0 + 8 * half + i) * NSEQ + m0 + rr] = acc[i] * 0.125f;
        }
    }
    __syncthreads();   // block-scope visibility of the strip (stores -> loads)

    // --- Phase B: masked softmax, wave w owns rows 2w, 2w+1 ---
#pragma unroll
    for (int rsel = 0; rsel < 2; ++rsel) {
        const int rl = w * 2 + rsel;
        const int n  = n0 + rl;
        if (n >= pd) {                       // fully-masked query row -> zeros
            for (int m = lane; m < NSEQ; m += 32) probs[rl][m] = (_Float16)0.0f;
        } else {
            const float* srow = sbase + (size_t)n * NSEQ;   // L2-resident window
            float vals[32];
            float mx = -3.4e38f;
#pragma unroll
            for (int i = 0; i < 32; ++i) {
                int m = lane + 32 * i;
                float vv = srow[m];
                vv = (m < pd) ? vv : -3.4e38f;
                vals[i] = vv;
                mx = fmaxf(mx, vv);
            }
#pragma unroll
            for (int off = 16; off >= 1; off >>= 1)
                mx = fmaxf(mx, __shfl_xor(mx, off, 32));
            float sum = 0.0f;
#pragma unroll
            for (int i = 0; i < 32; ++i) {
                int m = lane + 32 * i;
                float e = (m < pd) ? __expf(vals[i] - mx) : 0.0f;
                vals[i] = e;
                sum += e;
            }
#pragma unroll
            for (int off = 16; off >= 1; off >>= 1)
                sum += __shfl_xor(sum, off, 32);
            float inv = 1.0f / sum;
#pragma unroll
            for (int i = 0; i < 32; ++i)
                probs[rl][lane + 32 * i] = (_Float16)(vals[i] * inv);
        }
    }
    __syncthreads();

    // --- attention_values partial column sums (deterministic, no atomics) ---
    for (int m = threadIdx.x; m < NSEQ; m += 256) {
        float s = 0.0f;
#pragma unroll
        for (int r = 0; r < 16; ++r) s += (float)probs[r][m];
        avp[blk * NSEQ + m] = s;
    }

    // --- Phase C: context = probs @ V, waves 0..3 own one 16-wide d tile ---
    if (w < 4) {
        const int d0 = w * 16;
        const _Float16* vrow = v16t + ((size_t)bh * DH + d0 + rr) * NSEQ; // Vt[d][n]
        const _Float16* prow = &probs[rr][0];
        v8f acc = {};
#pragma unroll 4
        for (int m0 = 0; m0 < NSEQ; m0 += 32) {
            v16h a  = cat8(*(const v8h*)(prow + m0 + 8 * half),
                           *(const v8h*)(prow + m0 + 16 + 8 * half));
            v16h bm = cat8(*(const v8h*)(vrow + m0 + 8 * half),
                           *(const v8h*)(vrow + m0 + 16 + 8 * half));
            acc = __builtin_amdgcn_wmma_f32_16x16x32_f16(false, a, false, bm,
                                                         (short)0, acc, false, false);
        }
        float* ctx = dout + C_OFF + (size_t)bh * NSEQ * DH;
#pragma unroll
        for (int i = 0; i < 8; ++i)
            __builtin_nontemporal_store(acc[i],
                ctx + (n0 + 8 * half + i) * DH + d0 + rr);
    }
}

// ---------------------------------------------------------------------------
// Kernel 5: reduce 64 per-row-block partials -> attention_values (B,h,N)
// ---------------------------------------------------------------------------
__global__ void __launch_bounds__(256) k_avreduce(const float* __restrict__ avp,
                                                  float* __restrict__ av) {
    int idx = blockIdx.x * 256 + threadIdx.x;   // 0..65535
    if (idx >= BH * NSEQ) return;
    int bh = idx >> 10, m = idx & 1023;
    float s = 0.0f;
#pragma unroll 8
    for (int rb = 0; rb < 64; ++rb)
        s += avp[((bh << 6) + rb) * NSEQ + m];
    __builtin_nontemporal_store(s, av + idx);
}

// ---------------------------------------------------------------------------
extern "C" void kernel_launch(void* const* d_in, const int* in_sizes, int n_in,
                              void* d_out, int out_size, void* d_ws, size_t ws_size,
                              hipStream_t stream) {
    const float* X  = (const float*)d_in[0];
    const int*   pv = (const int*)d_in[1];
    const float* Wq = (const float*)d_in[2];
    const float* bq = (const float*)d_in[3];
    const float* Wk = (const float*)d_in[4];
    const float* bk = (const float*)d_in[5];
    const float* Wv = (const float*)d_in[6];
    const float* bv = (const float*)d_in[7];
    float* out = (float*)d_out;

    _Float16* ws16 = (_Float16*)d_ws;
    _Float16* X16  = ws16 + X16_OFF;
    _Float16* W16  = ws16 + W16_OFF;
    _Float16* q16  = ws16 + Q16_OFF;
    _Float16* k16  = ws16 + K16_OFF;
    _Float16* v16t = ws16 + V16T_OFF;
    int*   padd = (int*)((char*)d_ws + PADD_BYTE_OFF);
    float* avp  = (float*)((char*)d_ws + AVP_BYTE_OFF);

    // 1. convert f32 -> f16 operands
    k_convert<<<28672, 256, 0, stream>>>(X, Wq, Wk, Wv, X16, W16);
    // 2. pad boundary per batch
    k_padd<<<BATCH, 256, 0, stream>>>(pv, padd);
    // 3. QKV projections (WMMA)
    k_qkv<<<dim3(2048, 3), 256, 0, stream>>>(X16, W16, bq, bk, bv, out,
                                             q16, k16, v16t);
    // 4. fused scores -> softmax -> context (WMMA x2) + av partials
    k_sattn<<<BH * 64, 256, 0, stream>>>(q16, k16, v16t, padd, out, avp);
    // 5. attention_values reduction
    k_avreduce<<<256, 256, 0, stream>>>(avp, out + AV_OFF);
}
